// EncoderDecoderBlockFixed_71820443124368
// MI455X (gfx1250) — matrix operand order repaired and checked by reading, not compiled, not run on hardware
//
#include <hip/hip_runtime.h>

// ---------------- problem constants ----------------
#define B_   2
#define T_   2048
#define S_   2048
#define D_   768
#define H_   12
#define HD_  64
#define DFF_ 3072
#define HID_ 768

typedef __attribute__((ext_vector_type(16))) __bf16 v16bf;
typedef __attribute__((ext_vector_type(8)))  __bf16 v8bf;
typedef __attribute__((ext_vector_type(8)))  float  v8f;

union AF { v16bf v; __bf16 e[16]; };
union CF { v8f   v; float  e[8];  };

__device__ __forceinline__ v8f wmma_bf16(v16bf a, v16bf b, v8f c) {
  return __builtin_amdgcn_wmma_f32_16x16x32_bf16(false, a, false, b, (short)0, c, false, false);
}

// CDNA5 async global->LDS copy (ASYNCcnt tracked); 16B per lane, no VGPR round-trip.
__device__ __forceinline__ void async_b128(unsigned lds_off, const void* gptr) {
  asm volatile("global_load_async_to_lds_b128 %0, %1, off"
               :: "v"(lds_off), "v"(gptr) : "memory");
}
__device__ __forceinline__ void wait_async0() {
  asm volatile("s_wait_asynccnt 0x0" ::: "memory");
}
__device__ __forceinline__ unsigned lds_off(const void* p) {
  return (unsigned)(unsigned long long)p;   // low 32 bits of flat addr = LDS byte address
}

// ---------------- utility kernels ----------------
__global__ void cvt_bf16_kernel(const float* __restrict__ src, __bf16* __restrict__ dst, size_t n) {
  size_t i = (size_t)blockIdx.x * blockDim.x + threadIdx.x;
  if (i < n) dst[i] = (__bf16)src[i];
}

// src f32 [K][N] -> dst bf16 [N][K]  (32x32 LDS-tiled, both sides coalesced)
__global__ __launch_bounds__(256) void cvt_transpose_kernel(const float* __restrict__ src,
                                                            __bf16* __restrict__ dst,
                                                            int K, int N) {
  __shared__ float t[32][33];
  const int k0 = blockIdx.y * 32, n0 = blockIdx.x * 32;
  const int tx = threadIdx.x & 31, ty = threadIdx.x >> 5;   // ty in 0..7
#pragma unroll
  for (int i = ty; i < 32; i += 8) t[i][tx] = src[(size_t)(k0 + i) * N + n0 + tx];
  __syncthreads();
#pragma unroll
  for (int i = ty; i < 32; i += 8) dst[(size_t)(n0 + i) * K + k0 + tx] = (__bf16)t[tx][i];
}

__global__ void zero_kernel(float* __restrict__ dst, size_t n) {
  size_t i = (size_t)blockIdx.x * blockDim.x + threadIdx.x;
  if (i < n) dst[i] = 0.0f;
}

// position_bias[h,t,s] = rel_emb[rp[t,s], h]
__global__ void bias_kernel(const int* __restrict__ rp, const float* __restrict__ rel_emb,
                            float* __restrict__ out1) {
  size_t i = (size_t)blockIdx.x * blockDim.x + threadIdx.x;   // over T*T
  if (i >= (size_t)T_ * T_) return;
  int r = rp[i];
#pragma unroll
  for (int h = 0; h < H_; ++h)
    out1[(size_t)h * T_ * T_ + i] = rel_emb[r * H_ + h];
}

__global__ __launch_bounds__(256) void rmsnorm_bf16_kernel(const float* __restrict__ x,
                                                           const float* __restrict__ w,
                                                           __bf16* __restrict__ out) {
  __shared__ float red[256];
  size_t row = blockIdx.x;
  const float* xr = x + row * D_;
  float s = 0.f;
  for (int i = threadIdx.x; i < D_; i += 256) { float v = xr[i]; s += v * v; }
  red[threadIdx.x] = s;
  __syncthreads();
  for (int st = 128; st > 0; st >>= 1) {
    if (threadIdx.x < st) red[threadIdx.x] += red[threadIdx.x + st];
    __syncthreads();
  }
  float inv = rsqrtf(red[0] / (float)D_ + 1e-6f);
  for (int i = threadIdx.x; i < D_; i += 256)
    out[row * D_ + i] = (__bf16)(xr[i] * inv * w[i]);
}

// ---------------- generic bf16 WMMA GEMM (async double-buffered, B pre-transposed) ----
// C[M,N] = A[M,K] @ B[K,N], with B passed as Bt[N][K].
// EPI: 0 = f32 store, 1 = bf16 store, 2 = relu->bf16, 3 = f32 + res, 4 = bf16 transposed [N][M]
template<int EPI>
__global__ __launch_bounds__(128) void gemm_bf16_kernel(
    const __bf16* __restrict__ A, const __bf16* __restrict__ Bt,
    int M, int N, int K,
    const float* __restrict__ res, float* __restrict__ outF, __bf16* __restrict__ outB) {
  __shared__ __align__(16) __bf16 ldsA[2][64][40];    // A  tile rows  [m][k], 80B rows
  __shared__ __align__(16) __bf16 ldsBT[2][64][40];   // Bt tile rows  [n][k]
  __shared__ __align__(16) __bf16 ldsT[64][72];       // transpose bounce (EPI==4)

  const int tid  = threadIdx.x;
  const int lane = tid & 31;
  const int wave = tid >> 5;
  const int lr   = lane & 15;
  const int lh   = lane >> 4;

  const int bm = blockIdx.y * 64;
  const int bn = blockIdx.x * 64;
  const int wm = (wave >> 1) * 32;
  const int wn = (wave & 1) * 32;

  const int tRow = (tid * 16) >> 5, tCol = (tid * 16) & 31;   // 64x32 tile, 16 elems/thread

  CF acc[2][2];
#pragma unroll
  for (int i = 0; i < 2; ++i)
#pragma unroll
    for (int j = 0; j < 2; ++j)
#pragma unroll
      for (int r = 0; r < 8; ++r) acc[i][j].e[r] = 0.f;

  auto stage = [&](int buf, int k0) {
    const __bf16* as = A  + (size_t)(bm + tRow) * K + k0 + tCol;
    async_b128(lds_off(&ldsA[buf][tRow][tCol]),      as);
    async_b128(lds_off(&ldsA[buf][tRow][tCol + 8]),  as + 8);
    const __bf16* bs = Bt + (size_t)(bn + tRow) * K + k0 + tCol;
    async_b128(lds_off(&ldsBT[buf][tRow][tCol]),     bs);
    async_b128(lds_off(&ldsBT[buf][tRow][tCol + 8]), bs + 8);
  };

  stage(0, 0);
  wait_async0();
  __syncthreads();

  int buf = 0;
  for (int k0 = 0; k0 < K; k0 += 32) {
    if (k0 + 32 < K) stage(buf ^ 1, k0 + 32);   // overlap next tile with WMMAs

    AF af[2];
#pragma unroll
    for (int i = 0; i < 2; ++i) {
      int row = wm + i * 16 + lr;
      *(v8bf*)&af[i].e[0] = *(const v8bf*)&ldsA[buf][row][lh * 8];
      *(v8bf*)&af[i].e[8] = *(const v8bf*)&ldsA[buf][row][16 + lh * 8];
    }
    AF bf[2];
#pragma unroll
    for (int j = 0; j < 2; ++j) {
      int col = wn + j * 16 + lr;
      *(v8bf*)&bf[j].e[0] = *(const v8bf*)&ldsBT[buf][col][lh * 8];
      *(v8bf*)&bf[j].e[8] = *(const v8bf*)&ldsBT[buf][col][16 + lh * 8];
    }
#pragma unroll
    for (int i = 0; i < 2; ++i)
#pragma unroll
      for (int j = 0; j < 2; ++j)
        acc[i][j].v = wmma_bf16(af[i].v, bf[j].v, acc[i][j].v);

    wait_async0();
    __syncthreads();
    buf ^= 1;
  }

  if (EPI == 4) {
    // transposed bf16 store: out[N][M].  Bounce through LDS for coalesced stores.
#pragma unroll
    for (int i = 0; i < 2; ++i)
#pragma unroll
      for (int j = 0; j < 2; ++j)
#pragma unroll
        for (int r = 0; r < 8; ++r)
          ldsT[wn + j * 16 + lr][wm + i * 16 + lh * 8 + r] = (__bf16)acc[i][j].e[r];
    __syncthreads();
    int col = tid >> 1;
    int r0  = (tid & 1) * 32;
    __bf16* dstp = outB + (size_t)(bn + col) * M + bm + r0;
#pragma unroll
    for (int c = 0; c < 32; c += 8)
      *(v8bf*)(dstp + c) = *(const v8bf*)&ldsT[col][r0 + c];
  } else {
#pragma unroll
    for (int i = 0; i < 2; ++i)
#pragma unroll
      for (int j = 0; j < 2; ++j)
#pragma unroll
        for (int r = 0; r < 8; ++r) {
          int row = bm + wm + i * 16 + lh * 8 + r;
          int col = bn + wn + j * 16 + lr;
          size_t off = (size_t)row * N + col;
          float v = acc[i][j].e[r];
          if      (EPI == 1) outB[off] = (__bf16)v;
          else if (EPI == 2) outB[off] = (__bf16)fmaxf(v, 0.f);
          else if (EPI == 3) outF[off] = v + res[off];
          else               outF[off] = v;
        }
  }
}

// ---------------- flash attention (async double-buffered K / Vt chunks) ----------------
// q,k : [B*Tq|Skv][HID] bf16 ; vt : [HID][B*Skv] bf16 (transposed V)
template<bool HAS_BIAS>
__global__ __launch_bounds__(128) void attn_kernel(
    const __bf16* __restrict__ q, const __bf16* __restrict__ k, const __bf16* __restrict__ vt,
    const float* __restrict__ mask,   // [B,1,Tq,Skv]
    const float* __restrict__ bias,   // [H,Tq,Skv]
    __bf16* __restrict__ y, int Tq, int Skv, int MKV) {
  __shared__ __align__(16) __bf16 ldsK[2][32][72];    // [key][hd]
  __shared__ __align__(16) __bf16 ldsVT[2][64][40];   // [hd][key]
  __shared__ __align__(16) __bf16 ldsP[4][16][40];

  const int tid  = threadIdx.x;
  const int lane = tid & 31;
  const int wave = tid >> 5;
  const int lr   = lane & 15;
  const int lh   = lane >> 4;

  const int b  = blockIdx.z;
  const int h  = blockIdx.y;
  const int qw = blockIdx.x * 64 + wave * 16;

  const int cKey = (tid * 16) >> 6, cC = (tid * 16) & 63;   // K chunk 32x64
  const int vRow = (tid * 16) >> 5, vCol = (tid * 16) & 31; // Vt chunk 64x32

  // preload Q fragments (16 rows x 64 hd -> two K=32 fragments)
  AF qf[2];
  {
    const __bf16* qp = q + ((size_t)(b * Tq + qw + lr)) * HID_ + h * HD_;
#pragma unroll
    for (int kk = 0; kk < 2; ++kk) {
      *(v8bf*)&qf[kk].e[0] = *(const v8bf*)(qp + kk * 32 + lh * 8);
      *(v8bf*)&qf[kk].e[8] = *(const v8bf*)(qp + kk * 32 + 16 + lh * 8);
    }
  }

  float mrow[8], lrow[8];
  CF o[4];
#pragma unroll
  for (int r = 0; r < 8; ++r) { mrow[r] = -1e30f; lrow[r] = 0.f; }
#pragma unroll
  for (int n = 0; n < 4; ++n)
#pragma unroll
    for (int r = 0; r < 8; ++r) o[n].e[r] = 0.f;

  auto stage = [&](int buf, int s0) {
    const __bf16* ks = k + ((size_t)(b * Skv + s0 + cKey)) * HID_ + h * HD_ + cC;
    async_b128(lds_off(&ldsK[buf][cKey][cC]),      ks);
    async_b128(lds_off(&ldsK[buf][cKey][cC + 8]),  ks + 8);
    const __bf16* vs = vt + ((size_t)(h * HD_ + vRow)) * MKV + b * Skv + s0 + vCol;
    async_b128(lds_off(&ldsVT[buf][vRow][vCol]),     vs);
    async_b128(lds_off(&ldsVT[buf][vRow][vCol + 8]), vs + 8);
  };

  stage(0, 0);
  wait_async0();
  __syncthreads();

  int buf = 0;
  for (int s0 = 0; s0 < Skv; s0 += 32) {
    if (s0 + 32 < Skv) stage(buf ^ 1, s0 + 32);

    // scores: S[16 q][32 keys] = Q @ K^T
    CF sc[2];
#pragma unroll
    for (int j = 0; j < 2; ++j) {
#pragma unroll
      for (int r = 0; r < 8; ++r) sc[j].e[r] = 0.f;
#pragma unroll
      for (int kk = 0; kk < 2; ++kk) {
        AF bk;
        *(v8bf*)&bk.e[0] = *(const v8bf*)&ldsK[buf][j * 16 + lr][kk * 32 + lh * 8];
        *(v8bf*)&bk.e[8] = *(const v8bf*)&ldsK[buf][j * 16 + lr][kk * 32 + 16 + lh * 8];
        sc[j].v = wmma_bf16(qf[kk].v, bk.v, sc[j].v);
      }
    }
    // mask (+ optional position bias)
#pragma unroll
    for (int j = 0; j < 2; ++j)
#pragma unroll
      for (int r = 0; r < 8; ++r) {
        int t = qw + lh * 8 + r;
        int s = s0 + j * 16 + lr;
        float val = sc[j].e[r] + mask[((size_t)b * Tq + t) * Skv + s];
        if (HAS_BIAS) val += bias[((size_t)h * Tq + t) * Skv + s];
        sc[j].e[r] = val;
      }
    // online softmax (rows in 16-lane halves; xor masks 1..8 stay in-half)
#pragma unroll
    for (int r = 0; r < 8; ++r) {
      float mx = fmaxf(sc[0].e[r], sc[1].e[r]);
#pragma unroll
      for (int d = 8; d >= 1; d >>= 1) mx = fmaxf(mx, __shfl_xor(mx, d, 32));
      float mnew = fmaxf(mrow[r], mx);
      float p0 = __expf(sc[0].e[r] - mnew);
      float p1 = __expf(sc[1].e[r] - mnew);
      float scale = __expf(mrow[r] - mnew);
      float ps = p0 + p1;
#pragma unroll
      for (int d = 8; d >= 1; d >>= 1) ps += __shfl_xor(ps, d, 32);
      lrow[r] = lrow[r] * scale + ps;
      mrow[r] = mnew;
#pragma unroll
      for (int n = 0; n < 4; ++n) o[n].e[r] *= scale;
      sc[0].e[r] = p0;
      sc[1].e[r] = p1;
    }
    // C-layout -> A-layout via per-wave LDS tile
#pragma unroll
    for (int j = 0; j < 2; ++j)
#pragma unroll
      for (int r = 0; r < 8; ++r)
        ldsP[wave][lh * 8 + r][j * 16 + lr] = (__bf16)sc[j].e[r];
    AF pf;
    *(v8bf*)&pf.e[0] = *(const v8bf*)&ldsP[wave][lr][lh * 8];
    *(v8bf*)&pf.e[8] = *(const v8bf*)&ldsP[wave][lr][16 + lh * 8];
    // O += P @ V   (V stored transposed: contiguous along keys)
#pragma unroll
    for (int n = 0; n < 4; ++n) {
      AF bv;
      *(v8bf*)&bv.e[0] = *(const v8bf*)&ldsVT[buf][n * 16 + lr][lh * 8];
      *(v8bf*)&bv.e[8] = *(const v8bf*)&ldsVT[buf][n * 16 + lr][16 + lh * 8];
      o[n].v = wmma_bf16(pf.v, bv.v, o[n].v);
    }

    wait_async0();
    __syncthreads();
    buf ^= 1;
  }

#pragma unroll
  for (int n = 0; n < 4; ++n)
#pragma unroll
    for (int r = 0; r < 8; ++r) {
      int t = qw + lh * 8 + r;
      y[((size_t)(b * Tq + t)) * HID_ + h * HD_ + n * 16 + lr] = (__bf16)(o[n].e[r] / lrow[r]);
    }
}

// ---------------- host orchestration ----------------
template<int EPI>
static inline void gemm(const __bf16* A, const __bf16* Bt, int M, int N, int K,
                        const float* res, float* outF, __bf16* outB, hipStream_t st) {
  dim3 g(N / 64, M / 64), b(128);
  gemm_bf16_kernel<EPI><<<g, b, 0, st>>>(A, Bt, M, N, K, res, outF, outB);
}

extern "C" void kernel_launch(void* const* d_in, const int* in_sizes, int n_in,
                              void* d_out, int out_size, void* d_ws, size_t ws_size,
                              hipStream_t stream) {
  const float* x     = (const float*)d_in[0];
  const float* mask  = (const float*)d_in[1];
  const int*   rp    = (const int*)  d_in[2];
  const float* enc   = (const float*)d_in[3];
  const float* emask = (const float*)d_in[4];
  const float* sn_w  = (const float*)d_in[5];
  const float* wq    = (const float*)d_in[6];
  const float* wk    = (const float*)d_in[7];
  const float* wv    = (const float*)d_in[8];
  const float* wo    = (const float*)d_in[9];
  const float* rel   = (const float*)d_in[10];
  const float* cn_w  = (const float*)d_in[11];
  const float* cwq   = (const float*)d_in[12];
  const float* cwk   = (const float*)d_in[13];
  const float* cwv   = (const float*)d_in[14];
  const float* cwo   = (const float*)d_in[15];
  const float* pn_w  = (const float*)d_in[16];
  const float* ffwi  = (const float*)d_in[17];
  const float* ffwo  = (const float*)d_in[18];

  const size_t MT = (size_t)B_ * T_;                 // 4096 rows
  float* out0 = (float*)d_out;                       // [B,T,D]
  float* out1 = out0 + MT * D_;                      // [1,H,T,T]
  float* out2 = out1 + (size_t)H_ * T_ * T_;         // [B,1,T,S]

  char* ws = (char*)d_ws;
  size_t off = 0;
  auto take = [&](size_t bytes) { char* p = ws + off; off += (bytes + 255) & ~(size_t)255; return p; };
  __bf16* wq_t  = (__bf16*)take((size_t)D_ * HID_ * 2);   // all weights stored transposed [N][K]
  __bf16* wk_t  = (__bf16*)take((size_t)D_ * HID_ * 2);
  __bf16* wv_t  = (__bf16*)take((size_t)D_ * HID_ * 2);
  __bf16* wo_t  = (__bf16*)take((size_t)HID_ * D_ * 2);
  __bf16* cwq_t = (__bf16*)take((size_t)D_ * HID_ * 2);
  __bf16* cwk_t = (__bf16*)take((size_t)D_ * HID_ * 2);
  __bf16* cwv_t = (__bf16*)take((size_t)D_ * HID_ * 2);
  __bf16* cwo_t = (__bf16*)take((size_t)HID_ * D_ * 2);
  __bf16* wi_t  = (__bf16*)take((size_t)D_ * DFF_ * 2);
  __bf16* wo2_t = (__bf16*)take((size_t)DFF_ * D_ * 2);
  __bf16* h_b   = (__bf16*)take(MT * D_ * 2);
  __bf16* enc_b = (__bf16*)take(MT * D_ * 2);
  __bf16* q_b   = (__bf16*)take(MT * HID_ * 2);
  __bf16* k_b   = (__bf16*)take(MT * HID_ * 2);
  __bf16* vt_b  = (__bf16*)take(MT * HID_ * 2);            // V stored transposed [HID][MT]
  __bf16* y_b   = (__bf16*)take(MT * HID_ * 2);
  float*  attn  = (float*)take(MT * D_ * 4);
  __bf16* ff1   = (__bf16*)take(MT * DFF_ * 2);

  auto cvtT = [&](const float* s, __bf16* d, int K, int N) {
    cvt_transpose_kernel<<<dim3(N / 32, K / 32), dim3(256), 0, stream>>>(s, d, K, N);
  };

  // bias outputs
  bias_kernel<<<dim3(((size_t)T_ * T_ + 255) / 256), dim3(256), 0, stream>>>(rp, rel, out1);
  zero_kernel<<<dim3(((size_t)B_ * T_ * S_ + 255) / 256), dim3(256), 0, stream>>>(out2, (size_t)B_ * T_ * S_);

  // bf16 weights (transposed) + encoder states (row-major A operand)
  cvtT(wq, wq_t, D_, HID_);   cvtT(wk, wk_t, D_, HID_);
  cvtT(wv, wv_t, D_, HID_);   cvtT(wo, wo_t, HID_, D_);
  cvtT(cwq, cwq_t, D_, HID_); cvtT(cwk, cwk_t, D_, HID_);
  cvtT(cwv, cwv_t, D_, HID_); cvtT(cwo, cwo_t, HID_, D_);
  cvtT(ffwi, wi_t, D_, DFF_); cvtT(ffwo, wo2_t, DFF_, D_);
  cvt_bf16_kernel<<<dim3((unsigned)((MT * D_ + 255) / 256)), dim3(256), 0, stream>>>(enc, enc_b, MT * D_);

  // ---- self attention ----
  rmsnorm_bf16_kernel<<<dim3((unsigned)MT), dim3(256), 0, stream>>>(x, sn_w, h_b);
  gemm<1>(h_b, wq_t, (int)MT, HID_, D_, nullptr, nullptr, q_b, stream);
  gemm<1>(h_b, wk_t, (int)MT, HID_, D_, nullptr, nullptr, k_b, stream);
  gemm<4>(h_b, wv_t, (int)MT, HID_, D_, nullptr, nullptr, vt_b, stream);
  attn_kernel<true><<<dim3(T_ / 64, H_, B_), dim3(128), 0, stream>>>(
      q_b, k_b, vt_b, mask, out1, y_b, T_, T_, (int)MT);
  gemm<3>(y_b, wo_t, (int)MT, D_, HID_, x, attn, nullptr, stream);

  // ---- cross attention ----
  rmsnorm_bf16_kernel<<<dim3((unsigned)MT), dim3(256), 0, stream>>>(attn, cn_w, h_b);
  gemm<1>(h_b, cwq_t, (int)MT, HID_, D_, nullptr, nullptr, q_b, stream);
  gemm<1>(enc_b, cwk_t, (int)MT, HID_, D_, nullptr, nullptr, k_b, stream);
  gemm<4>(enc_b, cwv_t, (int)MT, HID_, D_, nullptr, nullptr, vt_b, stream);
  attn_kernel<false><<<dim3(T_ / 64, H_, B_), dim3(128), 0, stream>>>(
      q_b, k_b, vt_b, emask, nullptr, y_b, T_, S_, (int)MT);
  gemm<3>(y_b, cwo_t, (int)MT, D_, HID_, attn, attn, nullptr, stream);

  // ---- FFN ----
  rmsnorm_bf16_kernel<<<dim3((unsigned)MT), dim3(256), 0, stream>>>(attn, pn_w, h_b);
  gemm<2>(h_b, wi_t, (int)MT, DFF_, D_, nullptr, nullptr, ff1, stream);
  gemm<3>(ff1, wo2_t, (int)MT, D_, DFF_, attn, out0, nullptr, stream);
}
